// MultiHeadAttention_48756468744423
// MI455X (gfx1250) — matrix operand order repaired and checked by reading, not compiled, not run on hardware
//
#include <hip/hip_runtime.h>
#include <hip/hip_bf16.h>

#define DEV __device__ __forceinline__

typedef __attribute__((ext_vector_type(16))) __bf16 bf16x16;
typedef __attribute__((ext_vector_type(8))) float f32x8;

DEV unsigned short f2bf(float f) {
  return __builtin_bit_cast(unsigned short, static_cast<__bf16>(f));
}

union FragU {
  bf16x16 v;
  uint4 q[2];
  unsigned short s[16];
};

// A-fragment (16x32 bf16): two contiguous 8-element (16B) runs per lane.
DEV bf16x16 ld_frag_2x16B(const unsigned short* p0, const unsigned short* p1) {
  FragU f;
  f.q[0] = *(const uint4*)p0;
  f.q[1] = *(const uint4*)p1;
  return f.v;
}
// B-fragment (32x16 bf16): one contiguous 16-element (32B) run per lane.
DEV bf16x16 ld_frag_32B(const unsigned short* p) {
  FragU f;
  f.q[0] = *(const uint4*)p;
  f.q[1] = *(const uint4*)(p + 8);
  return f.v;
}

DEV f32x8 wmma_bf16(bf16x16 a, bf16x16 b, f32x8 c) {
  return __builtin_amdgcn_wmma_f32_16x16x32_bf16(
      false, a, false, b, (short)0, c, false, false);
}

// ---- fused DPP16 butterfly reduction steps: one VALU op per step ----------
// v_max_num_f32_dpp d, s, s  computes  d = max(dpp(s), s)  (DPP on src0 only)
#define DPP_STEP(FN, OPC, CTRL)                                             \
  DEV float FN(float x) {                                                   \
    float d;                                                                \
    asm(OPC " %0, %1, %1 " CTRL                                             \
            " row_mask:0xf bank_mask:0xf bound_ctrl:1"                      \
        : "=v"(d)                                                           \
        : "v"(x));                                                          \
    return d;                                                               \
  }
DPP_STEP(max_x1, "v_max_num_f32_dpp", "quad_perm:[1,0,3,2]")
DPP_STEP(max_x2, "v_max_num_f32_dpp", "quad_perm:[2,3,0,1]")
DPP_STEP(max_x7, "v_max_num_f32_dpp", "row_half_mirror")
DPP_STEP(max_xF, "v_max_num_f32_dpp", "row_mirror")
DPP_STEP(add_x1, "v_add_f32_dpp", "quad_perm:[1,0,3,2]")
DPP_STEP(add_x2, "v_add_f32_dpp", "quad_perm:[2,3,0,1]")
DPP_STEP(add_x7, "v_add_f32_dpp", "row_half_mirror")
DPP_STEP(add_xF, "v_add_f32_dpp", "row_mirror")
#undef DPP_STEP

DEV float red_max16(float v) { return max_xF(max_x7(max_x2(max_x1(v)))); }
DEV float red_sum16(float v) { return add_xF(add_x7(add_x2(add_x1(v)))); }

// ---- CDNA5 async global->LDS copy (ASYNCcnt-tracked) ----------------------
DEV void async_copy_b128(unsigned lds_byte_off, const void* gptr) {
  asm volatile("global_load_async_to_lds_b128 %0, %1, off"
               :
               : "v"(lds_byte_off), "v"((unsigned long long)gptr)
               : "memory");
}
DEV void wait_async0() { asm volatile("s_wait_asynccnt 0" ::: "memory"); }
DEV unsigned lds_off(const void* p) { return (unsigned)(uintptr_t)p; }

// ---- register-held staging tiles (prefetch pipeline) ----------------------
template <typename T, int NE> struct SReg;
template <int NE> struct SReg<float, NE> { float4 v[NE / 4]; };
template <int NE> struct SReg<unsigned short, NE> { uint4 v[NE / 8]; };

template <int NE> DEV void sload(SReg<float, NE>& r, const float* s) {
#pragma unroll
  for (int i = 0; i < NE / 4; ++i) r.v[i] = ((const float4*)s)[i];
}
template <int NE> DEV void sload(SReg<unsigned short, NE>& r, const unsigned short* s) {
#pragma unroll
  for (int i = 0; i < NE / 8; ++i) r.v[i] = ((const uint4*)s)[i];
}
template <int NE> DEV void sstore(const SReg<float, NE>& r, unsigned short* d) {
  const float* f = (const float*)&r;
#pragma unroll
  for (int h = 0; h < NE / 16; ++h) {
    FragU u;
#pragma unroll
    for (int i = 0; i < 16; ++i) u.s[i] = f2bf(f[h * 16 + i]);
    *(uint4*)(d + h * 16) = u.q[0];
    *(uint4*)(d + h * 16 + 8) = u.q[1];
  }
}
template <int NE> DEV void sstore(const SReg<unsigned short, NE>& r, unsigned short* d) {
#pragma unroll
  for (int i = 0; i < NE / 8; ++i) *(uint4*)(d + i * 8) = r.v[i];
}

DEV void store_out(float* C, size_t i, float v) { C[i] = v; }
DEV void store_out(unsigned short* C, size_t i, float v) { C[i] = f2bf(v); }

// ---------------------------------------------------------------------------
// C[M,N] = (A[M,K] @ W[N,K]^T + bias) * oscale.  M%128==N%128==K%32==0.
// 128x128 block tile, 8 waves in 4(M)x2(N) grid, wave = 2x4 WMMA tiles.
// BK=32 stages, double-buffered LDS: one barrier per stage; the LDS commit
// of stage s+1 sits after the WMMA block of stage s (global latency hidden).
// ---------------------------------------------------------------------------
template <typename AT, typename OT>
__global__ __launch_bounds__(256, 1) void gemm_xwT_bias(
    const AT* __restrict__ A, const float* __restrict__ W,
    const float* __restrict__ bias, OT* __restrict__ C,
    int M, int N, int K, float oscale) {
  constexpr int BK = 32, LP = 40;  // padded pitch -> conflict-free frag reads
  __shared__ unsigned short As[2][128 * LP];
  __shared__ unsigned short Ws[2][128 * LP];

  const int t = threadIdx.x;
  const int lane = t & 31, w = t >> 5;
  const int wm = w >> 1, wn = w & 1;
  const int hrow = lane & 15, half = lane >> 4;
  const int bm = blockIdx.y * 128, bn = blockIdx.x * 128;

  f32x8 acc[2][4] = {};

  const int srow = t >> 1, scg = (t & 1) * 16;  // staging: 2 threads/row
  const AT* aptr = A + (size_t)(bm + srow) * K + scg;
  const float* wptr = W + (size_t)(bn + srow) * K + scg;

  SReg<AT, 16> ra;
  SReg<float, 16> rw;
  sload(ra, aptr);
  sload(rw, wptr);
  sstore(ra, &As[0][srow * LP + scg]);
  sstore(rw, &Ws[0][srow * LP + scg]);
  __syncthreads();

  const int nst = K / BK;
  for (int s = 0; s < nst; ++s) {
    const int p = s & 1;
    const bool more = (s + 1) < nst;
    if (more) {  // issue next tile's global loads; consumed after compute
      sload(ra, aptr + (s + 1) * BK);
      sload(rw, wptr + (s + 1) * BK);
      if (s + 2 < nst) {
        __builtin_prefetch(aptr + (s + 2) * BK, 0, 0);
        __builtin_prefetch(wptr + (s + 2) * BK, 0, 0);
      }
    }
    bf16x16 af[2], bq[4];
#pragma unroll
    for (int mi = 0; mi < 2; ++mi) {
      const unsigned short* r = &As[p][(wm * 32 + mi * 16 + hrow) * LP];
      af[mi] = ld_frag_2x16B(r + half * 16, r + half * 16 + 8);
    }
#pragma unroll
    for (int ni = 0; ni < 4; ++ni) {
      const unsigned short* r = &Ws[p][(wn * 64 + ni * 16 + hrow) * LP];
      bq[ni] = ld_frag_32B(r + half * 16);
    }
#pragma unroll
    for (int mi = 0; mi < 2; ++mi)
#pragma unroll
      for (int ni = 0; ni < 4; ++ni)
        acc[mi][ni] = wmma_bf16(af[mi], bq[ni], acc[mi][ni]);
    if (more) {  // convert + commit next tile to the other buffer
      sstore(ra, &As[p ^ 1][srow * LP + scg]);
      sstore(rw, &Ws[p ^ 1][srow * LP + scg]);
    }
    __syncthreads();
  }

#pragma unroll
  for (int ni = 0; ni < 4; ++ni) {
    const int n = bn + wn * 64 + ni * 16 + hrow;
    const float bv = bias[n];
#pragma unroll
    for (int mi = 0; mi < 2; ++mi) {
#pragma unroll
      for (int r = 0; r < 8; ++r) {
        const int m = bm + wm * 32 + mi * 16 + r + 8 * half;  // C-layout row
        store_out(C, (size_t)m * N + n, (acc[mi][ni][r] + bv) * oscale);
      }
    }
  }
}

// ---------------------------------------------------------------------------
// Flash attention, bf16 q/k/v in [B,S,D_MODEL].  q pre-scaled by
// (1/sqrt(Dh))*log2(e) -> softmax in base 2 (v_exp_f32 is exp2).
// Block = (b, h, 128 q-rows); 8 waves x 16 rows.  KV blocks of 64,
// double-buffered K/V tiles, one workgroup barrier per KV block.
// K tile staged via global_load_async_to_lds_b128 (ASYNCcnt); V tile
// register-staged (needs transpose on commit); the wave-private P round
// trip is ordered by a wave-local s_wait_dscnt.
// ---------------------------------------------------------------------------
__global__ __launch_bounds__(256, 1) void mha_flash_attn(
    const unsigned short* __restrict__ qp, const unsigned short* __restrict__ kp,
    const unsigned short* __restrict__ vp, unsigned short* __restrict__ ctx) {
  constexpr int DM = 1024, S = 2048, DH = 64;
  constexpr int KP = 72, VP2 = 72, PP = 72;
  __shared__ unsigned short Ks[2][64 * KP];   // K tile [j][d]
  __shared__ unsigned short Vt[2][64 * VP2];  // V tile transposed [d][j]
  __shared__ unsigned short Ps[8 * 16 * PP];  // per-wave P tiles [m][j]

  const int t = threadIdx.x;
  const int lane = t & 31, w = t >> 5;
  const int hrow = lane & 15, half = lane >> 4;
  const int qb = blockIdx.x, h = blockIdx.y, b = blockIdx.z;
  const size_t headbase = (size_t)b * S * DM + (size_t)h * DH;
  const int qrow0 = qb * 128 + w * 16;

  // Q A-fragments straight from global (already scaled in projection).
  bf16x16 aq[2];
  {
    const unsigned short* qr = qp + headbase + (size_t)(qrow0 + hrow) * DM;
#pragma unroll
    for (int f = 0; f < 2; ++f)
      aq[f] = ld_frag_2x16B(qr + f * 32 + half * 16, qr + f * 32 + half * 16 + 8);
  }

  f32x8 oacc[4] = {};
  float mrun[8], lrun[8];
#pragma unroll
  for (int r = 0; r < 8; ++r) { mrun[r] = -1e30f; lrun[r] = 0.f; }

  unsigned short* Pw = &Ps[w * 16 * PP];
  const int sj = t >> 2, scg = (t & 3) * 16;  // staging: 4 threads/row
  const unsigned short* kb = kp + headbase + (size_t)sj * DM + scg;
  const unsigned short* vb = vp + headbase + (size_t)sj * DM + scg;

  // prologue: async-copy K block 0, register-stage + transpose V block 0
  {
    const unsigned kd = lds_off(&Ks[0][sj * KP + scg]);
    async_copy_b128(kd, kb);
    async_copy_b128(kd + 16, kb + 8);
  }
  FragU vrg;
  vrg.q[0] = *(const uint4*)vb;
  vrg.q[1] = *(const uint4*)(vb + 8);
#pragma unroll
  for (int e = 0; e < 16; ++e) Vt[0][(scg + e) * VP2 + sj] = vrg.s[e];
  wait_async0();
  __syncthreads();

  constexpr int NIT = S / 64;
  for (int it = 0; it < NIT; ++it) {
    const int p = it & 1;
    const bool more = (it + 1) < NIT;
    if (more) {  // async K copy into back buffer + V prefetch to registers
      const unsigned short* kn = kb + (size_t)(it + 1) * 64 * DM;
      const unsigned short* vn = vb + (size_t)(it + 1) * 64 * DM;
      const unsigned kd = lds_off(&Ks[p ^ 1][sj * KP + scg]);
      async_copy_b128(kd, kn);
      async_copy_b128(kd + 16, kn + 8);
      vrg.q[0] = *(const uint4*)vn;
      vrg.q[1] = *(const uint4*)(vn + 8);
    }

    // scores = q @ k^T  (contraction over d, 2 k-steps of 32)
    f32x8 cf[4] = {};
#pragma unroll
    for (int jt = 0; jt < 4; ++jt) {
      const unsigned short* kr = &Ks[p][(jt * 16 + hrow) * KP];
#pragma unroll
      for (int f = 0; f < 2; ++f) {
        bf16x16 bk = ld_frag_32B(kr + f * 32 + half * 16);
        cf[jt] = wmma_bf16(aq[f], bk, cf[jt]);
      }
    }

    // streaming softmax in base 2 (row m = r + 8*half, per 16-lane half)
#pragma unroll
    for (int r = 0; r < 8; ++r) {
      float mx = fmaxf(fmaxf(cf[0][r], cf[1][r]), fmaxf(cf[2][r], cf[3][r]));
      mx = red_max16(mx);
      const float mn = fmaxf(mrun[r], mx);
      const float alpha = __builtin_amdgcn_exp2f(mrun[r] - mn);
      float rs = 0.f;
#pragma unroll
      for (int jt = 0; jt < 4; ++jt) {
        float pe = __builtin_amdgcn_exp2f(cf[jt][r] - mn);
        cf[jt][r] = pe;
        rs += pe;
      }
      rs = red_sum16(rs);
      lrun[r] = lrun[r] * alpha + rs;
      mrun[r] = mn;
#pragma unroll
      for (int dt = 0; dt < 4; ++dt) oacc[dt][r] *= alpha;
    }

    // C-layout -> A-fragment layout via WAVE-PRIVATE LDS round trip.
#pragma unroll
    for (int jt = 0; jt < 4; ++jt)
#pragma unroll
      for (int r = 0; r < 8; ++r)
        Pw[(r + 8 * half) * PP + jt * 16 + hrow] = f2bf(cf[jt][r]);
    // wave-local ordering only: wait for this wave's DS stores to land
    asm volatile("s_wait_dscnt 0" ::: "memory");

    // O += P @ V  (contraction over j, 2 k-steps of 32)
    const unsigned short* pr = &Pw[hrow * PP];
#pragma unroll
    for (int dt = 0; dt < 4; ++dt) {
      const unsigned short* vr = &Vt[p][(dt * 16 + hrow) * VP2];
#pragma unroll
      for (int f = 0; f < 2; ++f) {
        bf16x16 ap = ld_frag_2x16B(pr + f * 32 + half * 16, pr + f * 32 + half * 16 + 8);
        bf16x16 bv = ld_frag_32B(vr + f * 32 + half * 16);
        oacc[dt] = wmma_bf16(ap, bv, oacc[dt]);
      }
    }

    if (more) {  // commit transposed V block to the other buffer
#pragma unroll
      for (int e = 0; e < 16; ++e) Vt[p ^ 1][(scg + e) * VP2 + sj] = vrg.s[e];
    }
    wait_async0();  // K async copies must be in LDS before the barrier
    __syncthreads();
  }

  // normalize (v_rcp) and write context (bf16 [B,S,D_MODEL])
#pragma unroll
  for (int r = 0; r < 8; ++r) {
    const int m = qrow0 + r + 8 * half;
    const float inv = __builtin_amdgcn_rcpf(lrun[r]);
    unsigned short* orow = ctx + headbase + (size_t)m * DM;
#pragma unroll
    for (int dt = 0; dt < 4; ++dt)
      orow[dt * 16 + hrow] = f2bf(oacc[dt][r] * inv);
  }
}

// ---------------------------------------------------------------------------
extern "C" void kernel_launch(void* const* d_in, const int* in_sizes, int n_in,
                              void* d_out, int out_size, void* d_ws, size_t ws_size,
                              hipStream_t stream) {
  (void)in_sizes; (void)n_in; (void)out_size; (void)ws_size;
  const float* Q  = (const float*)d_in[0];
  const float* K  = (const float*)d_in[1];
  const float* V  = (const float*)d_in[2];
  const float* Wq = (const float*)d_in[3];
  const float* bq = (const float*)d_in[4];
  const float* Wk = (const float*)d_in[5];
  const float* bk = (const float*)d_in[6];
  const float* Wv = (const float*)d_in[7];
  const float* bv = (const float*)d_in[8];
  const float* Wo = (const float*)d_in[9];
  const float* bo = (const float*)d_in[10];
  float* out = (float*)d_out;

  const int B = 4, S = 2048, DM = 1024;
  const size_t NE = (size_t)B * S * DM;
  unsigned short* ws = (unsigned short*)d_ws;
  unsigned short* qp = ws;            // bf16 q (pre-scaled) [B,S,DM]
  unsigned short* kp = ws + NE;       // bf16 k
  unsigned short* vp = ws + 2 * NE;   // bf16 v
  unsigned short* cx = ws + 3 * NE;   // bf16 attention output

  // fold (1/sqrt(Dh)) * log2(e) into q so attention softmax runs in base 2
  const float qscale = 0.125f * 1.44269504088896340736f;

  dim3 gg(DM / 128, (B * S) / 128, 1);
  gemm_xwT_bias<float, unsigned short><<<gg, 256, 0, stream>>>(Q, Wq, bq, qp, B * S, DM, DM, qscale);
  gemm_xwT_bias<float, unsigned short><<<gg, 256, 0, stream>>>(K, Wk, bk, kp, B * S, DM, DM, 1.0f);
  gemm_xwT_bias<float, unsigned short><<<gg, 256, 0, stream>>>(V, Wv, bv, vp, B * S, DM, DM, 1.0f);

  dim3 ga(S / 128, 16, B);
  mha_flash_attn<<<ga, 256, 0, stream>>>(qp, kp, vp, cx);

  gemm_xwT_bias<unsigned short, float><<<gg, 256, 0, stream>>>(cx, Wo, bo, out, B * S, DM, DM, 1.0f);
}
